// TorchNet_49383533969533
// MI455X (gfx1250) — matrix operand order repaired and checked by reading, not compiled, not run on hardware
//
#include <hip/hip_runtime.h>

#define BB 2048
#define NN 1024
#define KK 16
#define LL 9

#define TB 32            // batch columns per workgroup (LDS value tile width)
#define NODES_PER_WG 128 // 8 waves x 16 nodes

typedef __attribute__((ext_vector_type(16))) _Float16 v16h;
typedef __attribute__((ext_vector_type(8)))  float    v8f;

// Async global->LDS copy of 16 bytes (gfx1250 GLOBAL_LOAD_ASYNC_TO_LDS_B128).
// ldsoff = LDS byte offset (low 32 bits of generic LDS address), gaddr = global src.
__device__ __forceinline__ void async_copy_b128(unsigned ldsoff, const void* gaddr) {
  asm volatile("global_load_async_to_lds_b128 %0, %1, off"
               :: "v"(ldsoff), "v"(gaddr)
               : "memory");
}

__device__ __forceinline__ void wait_asynccnt0() {
#if __has_builtin(__builtin_amdgcn_s_wait_asynccnt)
  __builtin_amdgcn_s_wait_asynccnt(0);
#else
  asm volatile("s_wait_asynccnt 0x0" ::: "memory");
#endif
}

// ---------------------------------------------------------------------------
// Layer kernel: vout[n, b] = tanh( sum_k params[n,k] * vin[children[n,k], b] )
// vin/vout are [NN, BB] f16 ping-pong buffers in workspace.
// Each block: NODES_PER_WG nodes x TB batch cols. Whole previous layer
// (all NN rows, TB cols = 64 KB) staged in LDS via async DMA (CDNA5 big-LDS),
// so the gather is LDS-local. Each wave computes TWO 16x16 output tiles
// (batch halves) sharing one A operand: 16x v_wmma_f32_16x16x32_f16 per wave.
// ---------------------------------------------------------------------------
__global__ __launch_bounds__(256) void layer_kernel(
    const _Float16* __restrict__ vin,   // [NN, BB]
    _Float16*       __restrict__ vout,  // [NN, BB]
    const float*    __restrict__ params,   // [NN, KK] (this layer)
    const int*      __restrict__ children) // [NN, KK] (this layer)
{
  __shared__ alignas(16) _Float16 tile[NN * TB];            // 64 KB value tile
  __shared__ alignas(16) int      cbuf[NODES_PER_WG * KK];  // 8 KB child idx

  const int tid      = threadIdx.x;
  const int nodeBase = blockIdx.x * NODES_PER_WG;
  const int b0       = blockIdx.y * TB;

  // --- Async-stage value tile: NN rows x TB f16 cols (64 B/row = 4x b128) ---
  {
    const unsigned tile_off = (unsigned)(uintptr_t)tile;
    for (int ch = tid; ch < NN * 4; ch += 256) {
      const int row  = ch >> 2;
      const int part = ch & 3;
      async_copy_b128(tile_off + (unsigned)(ch * 16),
                      (const char*)(vin + (size_t)row * BB + b0) + part * 16);
    }
  }
  // --- Async-stage children stripe: NODES_PER_WG*KK ints = 8 KB contiguous ---
  {
    const unsigned cbuf_off = (unsigned)(uintptr_t)cbuf;
    const char* csrc = (const char*)(children + (size_t)nodeBase * KK);
    for (int ch = tid; ch < (NODES_PER_WG * KK * 4) / 16; ch += 256)
      async_copy_b128(cbuf_off + (unsigned)(ch * 16), csrc + ch * 16);
  }

  const int wave = tid >> 5;
  const int lane = tid & 31;
  const int ng   = wave * 16;       // node-group base within the stripe
  const int i16  = lane & 15;       // A row / B,D column index
  const int hi   = lane >> 4;       // half-wave selector

  // Per-lane A operand (params), loaded straight from global (read exactly once).
  // 16-bit A 16x32 layout: lane<16 holds row M=lane, k in {0..7, 16..23};
  // lane>=16 holds k {8..15, 24..31}. Chunk m uses k=0..15 for node 2m and
  // k=16..31 for node 2m+1: even node -> elements 0..7, odd -> elements 8..15.
  v16h afull = {};
  {
    const int node_loc = ng + i16;
    const int kbase    = hi * 8;          // which 8 params this lane carries
    const int place    = (i16 & 1) * 8;   // low half (even node) / high half (odd)
    const float* prow  = params + (size_t)(nodeBase + node_loc) * KK + kbase;
#pragma unroll
    for (int t = 0; t < 8; ++t)
      afull[place + t] = (_Float16)prow[t];
  }
  const v16h az = {};

  wait_asynccnt0();
  __syncthreads();

  v8f acc0 = {};   // batch cols b0 + 0..15
  v8f acc1 = {};   // batch cols b0 + 16..31
#pragma unroll
  for (int m = 0; m < 8; ++m) {
    // Block-diagonal A: only rows 2m, 2m+1 active this chunk (shared by both tiles).
    v16h a = (((i16 >> 1) == m) ? afull : az);

    // B operands: gathered child values. Element kk<8 carries child hi*8+kk of
    // node 2m, kk>=8 carries child hi*8+(kk-8) of node 2m+1; column = batch col i16 (+16).
    const int nb = ng + 2 * m;
    v16h bA = {};
    v16h bB = {};
#pragma unroll
    for (int kk = 0; kk < 16; ++kk) {
      const int child = cbuf[(nb + (kk >> 3)) * KK + hi * 8 + (kk & 7)];
      const int base  = child * TB + i16;   // second tile folds +16 into ds offset
      bA[kk] = tile[base];
      bB[kk] = tile[base + 16];
    }

    acc0 = __builtin_amdgcn_wmma_f32_16x16x32_f16(
        false, a, false, bA, (short)0, acc0, false, false);
    acc1 = __builtin_amdgcn_wmma_f32_16x16x32_f16(
        false, a, false, bB, (short)0, acc1, false, false);
  }

  // D layout: VGPR r -> row r (lanes 0..15) / row r+8 (lanes 16..31), col = i16.
#pragma unroll
  for (int r = 0; r < 8; ++r) {
    const int node_loc = ng + r + hi * 8;
    _Float16* orow = vout + (size_t)(nodeBase + node_loc) * BB + b0 + i16;
    orow[0]  = (_Float16)tanhf(acc0[r]);
    orow[16] = (_Float16)tanhf(acc1[r]);
  }
}

// ------------------- transpose X [B,N] f32 -> V [N,B] f16 -------------------
__global__ __launch_bounds__(256) void transpose_in(
    const float* __restrict__ X, _Float16* __restrict__ V)
{
  __shared__ float t[32][33];
  const int n0 = blockIdx.x * 32;
  const int b0 = blockIdx.y * 32;
  const int lx = threadIdx.x & 31, ly = threadIdx.x >> 5;
#pragma unroll
  for (int r = 0; r < 32; r += 8)
    t[ly + r][lx] = X[(size_t)(b0 + ly + r) * NN + n0 + lx];
  __syncthreads();
#pragma unroll
  for (int r = 0; r < 32; r += 8)
    V[(size_t)(n0 + ly + r) * BB + b0 + lx] = (_Float16)t[lx][ly + r];
}

// ------------------- transpose V [N,B] f16 -> out [B,N] f32 -----------------
__global__ __launch_bounds__(256) void transpose_out(
    const _Float16* __restrict__ V, float* __restrict__ out)
{
  __shared__ float t[32][33];
  const int b0 = blockIdx.x * 32;
  const int n0 = blockIdx.y * 32;
  const int lx = threadIdx.x & 31, ly = threadIdx.x >> 5;
#pragma unroll
  for (int r = 0; r < 32; r += 8)
    t[ly + r][lx] = (float)V[(size_t)(n0 + ly + r) * BB + b0 + lx];
  __syncthreads();
#pragma unroll
  for (int r = 0; r < 32; r += 8)
    out[(size_t)(b0 + ly + r) * NN + n0 + lx] = t[lx][ly + r];
}

// ---------------------------------------------------------------------------
extern "C" void kernel_launch(void* const* d_in, const int* in_sizes, int n_in,
                              void* d_out, int out_size, void* d_ws, size_t ws_size,
                              hipStream_t stream) {
  (void)in_sizes; (void)n_in; (void)out_size; (void)ws_size;
  const float* X        = (const float*)d_in[0];
  const float* params   = (const float*)d_in[1];
  const int*   children = (const int*)d_in[2];
  float*       out      = (float*)d_out;

  _Float16* v0 = (_Float16*)d_ws;                 // 4 MB
  _Float16* v1 = v0 + (size_t)NN * BB;            // 4 MB

  transpose_in<<<dim3(NN / 32, BB / 32), 256, 0, stream>>>(X, v0);

  _Float16* cur = v0;
  _Float16* nxt = v1;
  for (int l = 0; l < LL; ++l) {
    layer_kernel<<<dim3(NN / NODES_PER_WG, BB / TB), 256, 0, stream>>>(
        cur, nxt,
        params   + (size_t)l * NN * KK,
        children + (size_t)l * NN * KK);
    _Float16* tmp = cur; cur = nxt; nxt = tmp;
  }

  transpose_out<<<dim3(BB / 32, NN / 32), 256, 0, stream>>>(cur, out);
}
